// AFNOMix2D_70420283785863
// MI455X (gfx1250) — matrix operand order repaired and checked by reading, not compiled, not run on hardware
//
#include <hip/hip_runtime.h>
#include <hip/hip_bf16.h>

// ---------------------------------------------------------------------------
// AFNO 2D mixer for MI455X (gfx1250, wave32, WMMA).
//  k0: prep weights -> f16 WMMA-fragment layout, complex folded into real K=192
//  k1: mask + LayerNorm (writes x_ln into d_out)
//  k2: rfft along L (ortho 1/8), fp32, LDS tile + mod-64 twiddle table
//  k3: fft  along S (ortho 1/8), fp32, in place (sign=+1)
//  k4: block-diag complex MLP via v_wmma_f32_16x16x32_f16, GELU, soft-shrink,
//      in place on spectrum
//  k3: ifft along S (sign=-1)
//  k5: irfft along L + residual  out = x_ln + gamma*y (in place on d_out)
// ---------------------------------------------------------------------------

typedef _Float16 half_t;
typedef __attribute__((ext_vector_type(16))) _Float16 v16h;
typedef __attribute__((ext_vector_type(8)))  _Float16 v8h;
typedef __attribute__((ext_vector_type(4)))  _Float16 v4h;
typedef __attribute__((ext_vector_type(8)))  float    v8f;
typedef __attribute__((ext_vector_type(4)))  float    v4f;

#define DDIM   768
#define NBLK   8
#define BLKSZ  96
#define LHAT   33
#define TWO_PI 6.28318530717958647692f

// ---------------------------------------------------------------------------
// k0: pack weights into WMMA B-fragment-native layout.
// Matrices (per block k): W1R=[w1r;-w1i], W1I=[w1i;w1r], W2R=[w2r;-w2i],
// W2I=[w2i;w2r] each 192x96 f16, stored as [mat][k][ntile(6)][ktile(6)][512]
// with element (kk,col) at col*32 + (kk with bits 3,4 swapped) -> lane loads 16 contiguous halves.
// ---------------------------------------------------------------------------
__global__ void prep_weights_kernel(const float* __restrict__ w1,
                                    const float* __restrict__ w2,
                                    half_t* __restrict__ wp) {
  int idx = blockIdx.x * blockDim.x + threadIdx.x;
  if (idx >= 4 * NBLK * 36 * 512) return;
  int e   = idx & 511;
  int col = e >> 5;            // 0..15 (N within tile)
  int kk  = (e & 7) | (((e >> 4) & 1) << 3) | (((e >> 3) & 1) << 4);  // 0..31 (K within tile, table order per lane half)
  int st  = idx >> 9;
  int kt  = st % 6; st /= 6;
  int nt  = st % 6; st /= 6;
  int k   = st % NBLK;
  int mat = st / NBLK;         // 0=W1R 1=W1I 2=W2R 3=W2I
  int kap = kt * 32 + kk;      // 0..191  (K of [real|imag] concat)
  int j   = nt * 16 + col;     // 0..95   (output column)
  const float* w = (mat < 2) ? w1 : w2;
  int  hiK = (kap >= BLKSZ);
  int  i   = kap - (hiK ? BLKSZ : 0);
  float vr = w[((size_t)(0 * NBLK + k) * BLKSZ + i) * BLKSZ + j];
  float vi = w[((size_t)(1 * NBLK + k) * BLKSZ + i) * BLKSZ + j];
  float val = ((mat & 1) == 0) ? (hiK ? -vi : vr)   // real-output matrix
                               : (hiK ?  vr : vi);  // imag-output matrix
  wp[idx] = (half_t)val;
}

// ---------------------------------------------------------------------------
// k1: mask + LayerNorm, one wave32 per token (24 values/lane).
// ---------------------------------------------------------------------------
__global__ void ln_kernel(const float* __restrict__ x,
                          const unsigned char* __restrict__ pad,
                          const float* __restrict__ lnw,
                          const float* __restrict__ lnb,
                          float* __restrict__ xln) {
  int wave = threadIdx.x >> 5;
  int lane = threadIdx.x & 31;
  int t    = blockIdx.x * 8 + wave;
  const float* xp = x   + (size_t)t * DDIM;
  float*       op = xln + (size_t)t * DDIM;
  float msk = pad[t] ? 0.0f : 1.0f;
  float v[24];
  float sum = 0.0f;
#pragma unroll
  for (int i = 0; i < 24; ++i) { v[i] = msk * xp[lane + i * 32]; sum += v[i]; }
#pragma unroll
  for (int o = 16; o > 0; o >>= 1) sum += __shfl_xor(sum, o, 32);
  float mu = sum * (1.0f / 768.0f);
  float sq = 0.0f;
#pragma unroll
  for (int i = 0; i < 24; ++i) { float d = v[i] - mu; sq += d * d; }
#pragma unroll
  for (int o = 16; o > 0; o >>= 1) sq += __shfl_xor(sq, o, 32);
  float rstd = rsqrtf(sq * (1.0f / 768.0f) + 1e-5f);
#pragma unroll
  for (int i = 0; i < 24; ++i) {
    int d = lane + i * 32;
    op[d] = (v[i] - mu) * rstd * lnw[d] + lnb[d];
  }
}

// ---------------------------------------------------------------------------
// k2: rfft along L (64 -> 33 bins), ortho scale 1/8 folded into twiddles.
// One block per (b,s,d-chunk of 64). Twiddle index = (kl*l) mod 64.
// ---------------------------------------------------------------------------
__global__ void fftL_kernel(const float* __restrict__ xln,
                            float* __restrict__ Zr, float* __restrict__ Zi) {
  __shared__ float sx[64 * 64];
  __shared__ float tc[64], ts[64];
  int bs  = blockIdx.x / 12;
  int d0  = (blockIdx.x % 12) * 64;
  int tid = threadIdx.x;
  if (tid < 64) {
    float ang = TWO_PI * (float)tid * (1.0f / 64.0f);
    tc[tid] =  __cosf(ang) * 0.125f;
    ts[tid] = -__sinf(ang) * 0.125f;
  }
  for (int i = tid; i < 4096; i += 256) {
    int l = i >> 6, dd = i & 63;
    sx[i] = xln[((size_t)bs * 64 + l) * DDIM + d0 + dd];
  }
  __syncthreads();
  for (int o = tid; o < LHAT * 64; o += 256) {
    int kl = o >> 6, dd = o & 63;
    float ar = 0.0f, ai = 0.0f;
#pragma unroll 8
    for (int l = 0; l < 64; ++l) {
      float v = sx[l * 64 + dd];
      int   j = (kl * l) & 63;
      ar += v * tc[j];
      ai += v * ts[j];
    }
    size_t g = ((size_t)bs * LHAT + kl) * DDIM + d0 + dd;
    Zr[g] = ar; Zi[g] = ai;
  }
}

// ---------------------------------------------------------------------------
// k3: complex FFT along S, in place. sign=+1 forward, -1 inverse (ortho 1/8).
// One block per (b,kl,d-chunk of 64).
// ---------------------------------------------------------------------------
__global__ void fftS_kernel(float* __restrict__ Zr, float* __restrict__ Zi,
                            float sign) {
  __shared__ float sr[64 * 64], si[64 * 64];
  __shared__ float tc[64], ts[64];
  int idx   = blockIdx.x;
  int chunk = idx % 12; idx /= 12;
  int kl    = idx % LHAT; idx /= LHAT;
  int b     = idx;
  int d0    = chunk * 64;
  int tid   = threadIdx.x;
  if (tid < 64) {
    float ang = TWO_PI * (float)tid * (1.0f / 64.0f);
    tc[tid] = __cosf(ang) * 0.125f;
    ts[tid] = -sign * __sinf(ang) * 0.125f;
  }
  for (int i = tid; i < 4096; i += 256) {
    int s = i >> 6, dd = i & 63;
    size_t g = (((size_t)b * 64 + s) * LHAT + kl) * DDIM + d0 + dd;
    sr[i] = Zr[g]; si[i] = Zi[g];
  }
  __syncthreads();
  for (int o = tid; o < 4096; o += 256) {
    int ks = o >> 6, dd = o & 63;
    float ar = 0.0f, ai = 0.0f;
#pragma unroll 8
    for (int s = 0; s < 64; ++s) {
      int   j  = (ks * s) & 63;
      float c  = tc[j], sn = ts[j];
      float zr = sr[s * 64 + dd], zi = si[s * 64 + dd];
      ar += zr * c - zi * sn;
      ai += zi * c + zr * sn;
    }
    size_t g = (((size_t)b * 64 + ks) * LHAT + kl) * DDIM + d0 + dd;
    Zr[g] = ar; Zi[g] = ai;
  }
}

// ---------------------------------------------------------------------------
// k4: block-diagonal complex MLP with WMMA. 8 waves/WG, one 16-token m-tile
// per wave, blockIdx.y = block index k. In place on (Zr,Zi).
// ---------------------------------------------------------------------------
__device__ __forceinline__ float gelu_exact(float x) {
  return 0.5f * x * (1.0f + erff(x * 0.70710678118654752f));
}

__global__ void __launch_bounds__(256)
mlp_kernel(float* __restrict__ Zr, float* __restrict__ Zi,
           const half_t* __restrict__ wp,
           const float* __restrict__ b1, const float* __restrict__ b2) {
  __shared__ half_t sH[8][16 * 192];   // hidden [Hr|Hi], 6KB per wave

  int wave = threadIdx.x >> 5;
  int lane = threadIdx.x & 31;
  int m    = lane & 15;                // row / column within fragment
  int hi   = lane >> 4;                // half-wave select (wave32 layout)
  int k    = blockIdx.y;               // block 0..7
  int t0   = (blockIdx.x * 8 + wave) * 16;

  // B-fragment: 32B-aligned v16h straight from prepacked global (L2-resident)
  auto loadB = [&](int mat, int nt, int kt) -> v16h {
    const half_t* p = wp + ((((size_t)mat * NBLK + k) * 36 + nt * 6 + kt) << 9)
                         + (m << 5) + hi * 16;
    return *(const v16h*)p;
  };

  // ---- layer 1:  A = [zr | zi] (16x192 f16) -------------------------------
  v8f accR[6], accI[6];
#pragma unroll
  for (int n = 0; n < 6; ++n)
#pragma unroll
    for (int e = 0; e < 8; ++e) { accR[n][e] = 0.0f; accI[n][e] = 0.0f; }

  for (int kt = 0; kt < 6; ++kt) {
    const float* src = (kt < 3) ? Zr : Zi;
    int off = (kt < 3) ? kt * 32 : (kt - 3) * 32;
    const float* p = src + (size_t)(t0 + m) * DDIM + k * BLKSZ + off + hi * 8;
    const v4f* q = (const v4f*)p;                 // 16B aligned
    v4h h0 = __builtin_convertvector(q[0], v4h);  // K = base+0..3
    v4h h1 = __builtin_convertvector(q[1], v4h);  // K = base+4..7
    v4h h2 = __builtin_convertvector(q[4], v4h);  // K = base+16..19
    v4h h3 = __builtin_convertvector(q[5], v4h);  // K = base+20..23
    v8h lo  = __builtin_shufflevector(h0, h1, 0, 1, 2, 3, 4, 5, 6, 7);
    v8h hi8 = __builtin_shufflevector(h2, h3, 0, 1, 2, 3, 4, 5, 6, 7);
    v16h a  = __builtin_shufflevector(lo, hi8, 0, 1, 2, 3, 4, 5, 6, 7,
                                      8, 9, 10, 11, 12, 13, 14, 15);
#pragma unroll
    for (int nt = 0; nt < 6; ++nt) {
      v16h bR = loadB(0, nt, kt);
      accR[nt] = __builtin_amdgcn_wmma_f32_16x16x32_f16(
          false, a, false, bR, (short)0, accR[nt], false, false);
      v16h bI = loadB(1, nt, kt);
      accI[nt] = __builtin_amdgcn_wmma_f32_16x16x32_f16(
          false, a, false, bI, (short)0, accI[nt], false, false);
    }
  }

  // bias + exact GELU; stage hidden to LDS (C-layout -> A-layout transpose)
#pragma unroll
  for (int nt = 0; nt < 6; ++nt) {
    int j = nt * 16 + m;
    float br = b1[(0 * NBLK + k) * BLKSZ + j];
    float bi = b1[(1 * NBLK + k) * BLKSZ + j];
#pragma unroll
    for (int e = 0; e < 8; ++e) {
      int row = e + hi * 8;
      sH[wave][row * 192 + j]          = (half_t)gelu_exact(accR[nt][e] + br);
      sH[wave][row * 192 + BLKSZ + j]  = (half_t)gelu_exact(accI[nt][e] + bi);
    }
  }
  __syncthreads();

  // ---- layer 2:  A = [Hr | Hi] from LDS -----------------------------------
  v8f oR[6], oI[6];
#pragma unroll
  for (int n = 0; n < 6; ++n)
#pragma unroll
    for (int e = 0; e < 8; ++e) { oR[n][e] = 0.0f; oI[n][e] = 0.0f; }

  for (int kt = 0; kt < 6; ++kt) {
    const half_t* hp = &sH[wave][m * 192 + kt * 32 + hi * 8];   // 16B aligned
    v8h lo  = *(const v8h*)hp;
    v8h hi8 = *(const v8h*)(hp + 16);
    v16h a  = __builtin_shufflevector(lo, hi8, 0, 1, 2, 3, 4, 5, 6, 7,
                                      8, 9, 10, 11, 12, 13, 14, 15);
#pragma unroll
    for (int nt = 0; nt < 6; ++nt) {
      v16h bR = loadB(2, nt, kt);
      oR[nt] = __builtin_amdgcn_wmma_f32_16x16x32_f16(
          false, a, false, bR, (short)0, oR[nt], false, false);
      v16h bI = loadB(3, nt, kt);
      oI[nt] = __builtin_amdgcn_wmma_f32_16x16x32_f16(
          false, a, false, bI, (short)0, oI[nt], false, false);
    }
  }

  // bias + complex soft-shrink, store in place
#pragma unroll
  for (int nt = 0; nt < 6; ++nt) {
    int j = nt * 16 + m;
    float br = b2[(0 * NBLK + k) * BLKSZ + j];
    float bi = b2[(1 * NBLK + k) * BLKSZ + j];
#pragma unroll
    for (int e = 0; e < 8; ++e) {
      int   row = e + hi * 8;
      float r   = oR[nt][e] + br;
      float im  = oI[nt][e] + bi;
      float mag = sqrtf(r * r + im * im);
      float sc  = fmaxf(mag - 0.01f, 0.0f) / (mag + 1e-9f);
      size_t g  = (size_t)(t0 + row) * DDIM + k * BLKSZ + j;
      Zr[g] = r * sc;
      Zi[g] = im * sc;
    }
  }
}

// ---------------------------------------------------------------------------
// k5: irfft along L (hermitian expansion) + residual. In place on d_out.
// ---------------------------------------------------------------------------
__global__ void ifftL_kernel(const float* __restrict__ Zr,
                             const float* __restrict__ Zi,
                             const unsigned char* __restrict__ pad,
                             const float* __restrict__ gamma,
                             float* __restrict__ out /* holds x_ln */) {
  __shared__ float sr[LHAT * 64], si[LHAT * 64];
  __shared__ float tc[64], ts[64];
  int bs  = blockIdx.x / 12;
  int d0  = (blockIdx.x % 12) * 64;
  int tid = threadIdx.x;
  if (tid < 64) {
    float ang = TWO_PI * (float)tid * (1.0f / 64.0f);
    tc[tid] = __cosf(ang) * 0.125f;
    ts[tid] = __sinf(ang) * 0.125f;
  }
  for (int i = tid; i < LHAT * 64; i += 256) {
    int kl = i >> 6, dd = i & 63;
    size_t g = ((size_t)bs * LHAT + kl) * DDIM + d0 + dd;
    sr[i] = Zr[g]; si[i] = Zi[g];
  }
  __syncthreads();
  for (int o = tid; o < 4096; o += 256) {
    int l = o >> 6, dd = o & 63;
    float acc = (sr[dd] + ((l & 1) ? -sr[32 * 64 + dd] : sr[32 * 64 + dd])) * 0.125f;
#pragma unroll 8
    for (int kl = 1; kl < 32; ++kl) {
      int j = (kl * l) & 63;
      acc += 2.0f * (sr[kl * 64 + dd] * tc[j] - si[kl * 64 + dd] * ts[j]);
    }
    float y = pad[bs * 64 + l] ? 0.0f : acc;
    size_t g = ((size_t)bs * 64 + l) * DDIM + d0 + dd;
    out[g] = out[g] + gamma[d0 + dd] * y;
  }
}

// ---------------------------------------------------------------------------
extern "C" void kernel_launch(void* const* d_in, const int* in_sizes, int n_in,
                              void* d_out, int out_size, void* d_ws, size_t ws_size,
                              hipStream_t stream) {
  const float*         x     = (const float*)d_in[0];
  const unsigned char* pad   = (const unsigned char*)d_in[1];
  const float*         lnw   = (const float*)d_in[2];
  const float*         lnb   = (const float*)d_in[3];
  const float*         gamma = (const float*)d_in[4];
  const float*         w1    = (const float*)d_in[5];
  const float*         b1    = (const float*)d_in[6];
  const float*         w2    = (const float*)d_in[7];
  const float*         b2    = (const float*)d_in[8];
  float*               out   = (float*)d_out;

  // workspace: spectrum (2 x 49.5 MiB) + packed f16 weights (1.2 MiB)
  const size_t specN = (size_t)8 * 64 * LHAT * DDIM;        // 12,976,128 f32
  char*   ws = (char*)d_ws;
  float*  Zr = (float*)ws;
  float*  Zi = (float*)(ws + specN * sizeof(float));
  half_t* wp = (half_t*)(ws + 2 * specN * sizeof(float));

  // 0) pack weights for WMMA
  int wElems = 4 * NBLK * 36 * 512;
  prep_weights_kernel<<<(wElems + 255) / 256, 256, 0, stream>>>(w1, w2, wp);
  // 1) mask + LayerNorm  (x_ln lives in d_out)
  ln_kernel<<<4096, 256, 0, stream>>>(x, pad, lnw, lnb, out);
  // 2) rfft along L
  fftL_kernel<<<8 * 64 * 12, 256, 0, stream>>>(out, Zr, Zi);
  // 3) fft along S (forward)
  fftS_kernel<<<8 * LHAT * 12, 256, 0, stream>>>(Zr, Zi, 1.0f);
  // 4) block-diagonal complex MLP (WMMA), in place on spectrum
  dim3 mg(132, NBLK);
  mlp_kernel<<<mg, 256, 0, stream>>>(Zr, Zi, wp, b1, b2);
  // 5) ifft along S (inverse)
  fftS_kernel<<<8 * LHAT * 12, 256, 0, stream>>>(Zr, Zi, -1.0f);
  // 6) irfft along L + residual (in place on d_out)
  ifftL_kernel<<<8 * 64 * 12, 256, 0, stream>>>(Zr, Zi, pad, gamma, out);
}
